// CapsuleRacingNetwork_47382079209558
// MI455X (gfx1250) — matrix operand order repaired and checked by reading, not compiled, not run on hardware
//
#include <hip/hip_runtime.h>
#include <math.h>

typedef __attribute__((ext_vector_type(16))) _Float16 v16h;
typedef __attribute__((ext_vector_type(8)))  float    v8f;

// ---------------- problem constants ----------------
#define BB      8
#define HIN     120
#define WIN     160
#define CIN     3
#define HOUT    56
#define WOUT    76
#define CCH     256                 // conv output channels
#define KREAL   243                 // 9*9*3 im2col depth
#define KPAD    256
#define KC      (KPAD/32)           // 8 K-chunks of 32
#define PIXB    (HOUT*WOUT)         // 4256 output pixels per batch
#define PIX     (BB*PIXB)           // 34048 total output pixels
#define MT      (PIX/16)            // 2128 M-tiles of 16 pixels
#define NROWS   (PIXB*16)           // 68096 capsule rows per batch
#define CAPS_F  (PIX*CCH)           // caps element count (8,716,288 f32)

// ws layout (bytes)
#define OFF_APACK 0u
#define OFF_CAPS  17432576u         // Apack = MT*8*32*16 halves = 17,432,576 B
#define OFF_BPACK 52297728u         // caps  = 34,865,152 B
#define OFF_CSUM  52428800u         // Bpack = 131,072 B
#define OFF_ACUM  52429312u         // colsum 512 B
#define OFF_Z     52437504u         // Acum 8192 B
#define OFF_GNUM  52438016u         // Z 512 B, Gnum 8192 B (contiguous with Z)

// ---------------- pack im2col A into WMMA f16 A-fragment layout ----------------
// Apack[((tile*8+kc)*32+lane)*16 + j]; A 16x32 f16 layout:
//   row = lane&15, hi = lane>>4, K = (j<8 ? j+8*hi : j+8+8*hi) + kc*32
__global__ __launch_bounds__(256) void caps_pack_a(const float* __restrict__ in,
                                                   _Float16* __restrict__ Apack) {
    unsigned idx = blockIdx.x * 256u + threadIdx.x;     // exactly MT*8*32*16 threads
    unsigned j    =  idx        & 15u;
    unsigned lane = (idx >> 4)  & 31u;
    unsigned kc   = (idx >> 9)  & 7u;
    unsigned tile =  idx >> 12;
    unsigned row  = lane & 15u, hi = lane >> 4;
    unsigned kl   = (j < 8u) ? (j + 8u * hi) : (j + 8u + 8u * hi);
    unsigned kg   = kc * 32u + kl;
    float v = 0.0f;
    if (kg < KREAL) {
        unsigned pix = tile * 16u + row;
        unsigned b   = pix / PIXB;
        unsigned pr  = pix - b * PIXB;
        unsigned oy  = pr / WOUT, ox = pr - oy * WOUT;
        unsigned ci  = kg % 3u, kxy = kg / 3u;
        unsigned kx  = kxy % 9u, ky = kxy / 9u;
        unsigned iy  = oy * 2u + ky, ix = ox * 2u + kx;
        v = in[((b * HIN + iy) * WIN + ix) * CIN + ci];
    }
    Apack[idx] = (_Float16)v;
}

// ---------------- pack conv weights into WMMA f16 B-fragment layout ----------------
// Bpack[((kc*16+nt)*32+lane)*16 + j]; B 32x16 f16: col = lane&15, K = (j<8 ? j+8*(lane>>4) : j+8+8*(lane>>4))
__global__ __launch_bounds__(256) void caps_pack_w(const float* __restrict__ w,
                                                   _Float16* __restrict__ Bpack) {
    unsigned idx = blockIdx.x * 256u + threadIdx.x;     // 8*16*32*16 = 65536 threads
    unsigned j    =  idx        & 15u;
    unsigned lane = (idx >> 4)  & 31u;
    unsigned nt   = (idx >> 9)  & 15u;
    unsigned kc   =  idx >> 13;
    unsigned kg   = kc * 32u + ((j < 8u) ? (j + 8u * (lane >> 4)) : (j + 8u + 8u * (lane >> 4)));
    unsigned ch   = nt * 16u + (lane & 15u);
    float v = (kg < KREAL) ? w[kg * CCH + ch] : 0.0f;
    Bpack[idx] = (_Float16)v;
}

// ---------------- conv GEMM (WMMA) + bias + relu + squash -> caps ----------------
// block = 4 waves; block owns one 16-pixel M-tile x all 256 channels.
// wave w handles capsules ct = 4w..4w+3 (16 channels each).
__global__ __launch_bounds__(128) void caps_conv_wmma(const _Float16* __restrict__ Apack,
                                                      const _Float16* __restrict__ Bpack,
                                                      const float* __restrict__ bias,
                                                      float* __restrict__ caps) {
    const unsigned tile = blockIdx.x;
    const unsigned tid  = threadIdx.x;
    const unsigned w    = tid >> 5;
    const unsigned lane = tid & 31u;
    const unsigned hi   = lane >> 4;
    const unsigned lo   = lane & 15u;

    const v16h* Ap = (const v16h*)(Apack + (size_t)tile * 4096u);
    const v16h* Bp = (const v16h*)Bpack;

    v8f acc[4] = {};
#pragma unroll
    for (int kc = 0; kc < KC; ++kc) {
        v16h a = Ap[kc * 32 + lane];
        if (kc + 1 < KC)
            __builtin_prefetch(&Ap[(kc + 1) * 32 + lane], 0, 1);   // global_prefetch_b8
#pragma unroll
        for (int j = 0; j < 4; ++j) {
            unsigned nt = w * 4u + j;
            v16h b = Bp[(kc * 16u + nt) * 32u + lane];
            acc[j] = __builtin_amdgcn_wmma_f32_16x16x32_f16(
                false, a, false, b, (short)0, acc[j], false, false);
        }
    }

    // epilogue: bias + relu + squash(per 16-lane half), store caps (= y layout)
#pragma unroll
    for (int j = 0; j < 4; ++j) {
        unsigned ct = w * 4u + j;
        unsigned ch = ct * 16u + lo;
        float bs = bias[ch];
#pragma unroll
        for (int v = 0; v < 8; ++v) {
            float val = acc[j][v] + bs;
            val = fmaxf(val, 0.0f);
            float sq = val * val;
            sq += __shfl_xor(sq, 1, 32);
            sq += __shfl_xor(sq, 2, 32);
            sq += __shfl_xor(sq, 4, 32);
            sq += __shfl_xor(sq, 8, 32);
            float scale = sq / ((1.0f + sq) * sqrtf(sq + 1e-9f));
            float out = val * scale;
            unsigned pix = tile * 16u + (unsigned)v + 8u * hi;
            caps[(size_t)pix * CCH + ch] = out;
        }
    }
}

// ---------------- column sums of caps: colsum[b][p] = sum_n caps[b,n,p] ----------------
__global__ __launch_bounds__(256) void caps_colsum(const float* __restrict__ caps,
                                                   float* __restrict__ colsum) {
    __shared__ float red[256];
    const unsigned b = blockIdx.y;
    const unsigned t = threadIdx.x;
    size_t base = (size_t)b * (size_t)(NROWS * 16) + (size_t)blockIdx.x * 4096u;
    float acc = 0.0f;
#pragma unroll
    for (int i = 0; i < 16; ++i) acc += caps[base + (size_t)i * 256u + t];
    red[t] = acc;
    __syncthreads();
    if (t < 16u) {
        float s = 0.0f;
#pragma unroll
        for (int g = 0; g < 16; ++g) s += red[g * 16 + t];
        atomicAdd(&colsum[b * 16u + t], s);
    }
}

// ---------------- logits + softmax-weighted accumulation (one routing iteration) ----
// thread (k = t&15, slot = t>>4) loops 32 rows; accumulates Z[b][k], Gnum[b][k][p].
__global__ __launch_bounds__(256) void caps_logits(const float* __restrict__ caps,
                                                   const float* __restrict__ Acum,
                                                   float* __restrict__ Z,
                                                   float* __restrict__ Gnum) {
    __shared__ float As[256];
    __shared__ float Zs[256];
    __shared__ float Gs[4096];
    const unsigned b = blockIdx.y;
    const unsigned t = threadIdx.x;
    const unsigned k = t & 15u, slot = t >> 4;
    As[t] = Acum[b * 256u + t];
    __syncthreads();

    const unsigned rowBase = blockIdx.x * 512u;
    float zacc = 0.0f;
    float gacc[16];
#pragma unroll
    for (int p = 0; p < 16; ++p) gacc[p] = 0.0f;

    for (int i = 0; i < 32; ++i) {
        unsigned n = rowBase + slot + 16u * (unsigned)i;
        const float* cr = caps + ((size_t)b * NROWS + n) * 16u;
        float c0[16];
#pragma unroll
        for (int p = 0; p < 16; ++p) c0[p] = cr[p];
        float tt = 0.0f;
#pragma unroll
        for (int p = 0; p < 16; ++p) tt = fmaf(c0[p], As[p * 16 + k], tt);
        float e = __expf(tt);         // logits bounded (<~1.6): safe without max-sub
        zacc += e;
#pragma unroll
        for (int p = 0; p < 16; ++p) gacc[p] = fmaf(e, c0[p], gacc[p]);
    }

    Zs[t] = zacc;
#pragma unroll
    for (int p = 0; p < 16; ++p) Gs[t * 16 + p] = gacc[p];
    __syncthreads();

    if (t < 16u) {
        float s = 0.0f;
#pragma unroll
        for (int g = 0; g < 16; ++g) s += Zs[g * 16 + t];
        atomicAdd(&Z[b * 16u + t], s);
    }
    {
        unsigned k2 = t >> 4, p2 = t & 15u;
        float s = 0.0f;
#pragma unroll
        for (int g = 0; g < 16; ++g) s += Gs[(g * 16 + k2) * 16 + p2];
        atomicAdd(&Gnum[(b * 16u + k2) * 16u + p2], s);
    }
}

// ---------------- tiny routing step: s = (G/Z)@W, v = squash(s), Acum += A(v) ------
// mode 0: iteration 0 (uniform c: G = colsum, Z = NROWS; Acum overwritten)
// mode 1: middle iteration (Acum += )      mode 2: last (write v to d_out)
__global__ __launch_bounds__(128) void caps_routing_step(const float* __restrict__ rw,
                                                         const float* __restrict__ colsum,
                                                         const float* __restrict__ Gnum,
                                                         const float* __restrict__ Z,
                                                         float* __restrict__ Acum,
                                                         float* __restrict__ out,
                                                         int mode) {
    __shared__ float Ws[4096];
    const unsigned t = threadIdx.x;             // 128 threads = (b,k) pairs
#pragma unroll
    for (int i = 0; i < 32; ++i) Ws[t + 128 * i] = rw[t + 128 * i];
    __syncthreads();

    const unsigned b = t >> 4, k = t & 15u;
    float Gp[16];
    float z;
    if (mode == 0) {
#pragma unroll
        for (int p = 0; p < 16; ++p) Gp[p] = colsum[b * 16u + p];
        z = (float)NROWS;
    } else {
#pragma unroll
        for (int p = 0; p < 16; ++p) Gp[p] = Gnum[(b * 16u + k) * 16u + p];
        z = Z[b * 16u + k];
    }
    float inv = 1.0f / z;

    float s[16];
#pragma unroll
    for (int l = 0; l < 16; ++l) {
        float a = 0.0f;
#pragma unroll
        for (int p = 0; p < 16; ++p) a = fmaf(Gp[p], Ws[p * 256 + k * 16 + l], a);
        s[l] = a * inv;
    }
    float sn = 0.0f;
#pragma unroll
    for (int l = 0; l < 16; ++l) sn = fmaf(s[l], s[l], sn);
    float scale = sn / ((1.0f + sn) * sqrtf(sn + 1e-9f));
    float v[16];
#pragma unroll
    for (int l = 0; l < 16; ++l) v[l] = s[l] * scale;

    if (mode == 2) {
#pragma unroll
        for (int l = 0; l < 16; ++l) out[b * 256u + k * 16u + l] = v[l];
    } else {
#pragma unroll
        for (int p = 0; p < 16; ++p) {
            float a = 0.0f;
#pragma unroll
            for (int l = 0; l < 16; ++l) a = fmaf(Ws[p * 256 + k * 16 + l], v[l], a);
            unsigned ai = b * 256u + p * 16u + k;
            if (mode == 0) Acum[ai] = a;
            else           Acum[ai] += a;
        }
    }
}

// ---------------- launch ----------------
extern "C" void kernel_launch(void* const* d_in, const int* in_sizes, int n_in,
                              void* d_out, int out_size, void* d_ws, size_t ws_size,
                              hipStream_t stream) {
    const float* inputs = (const float*)d_in[0];   // (8,120,160,3)
    const float* conv_w = (const float*)d_in[1];   // (9,9,3,256)
    const float* conv_b = (const float*)d_in[2];   // (256,)
    const float* rw     = (const float*)d_in[3];   // (16,16,16)
    float* out = (float*)d_out;                    // (8,16,16)

    char* ws = (char*)d_ws;
    _Float16* Apack  = (_Float16*)(ws + OFF_APACK);
    float*    caps   = (float*)   (ws + OFF_CAPS);
    _Float16* Bpack  = (_Float16*)(ws + OFF_BPACK);
    float*    colsum = (float*)   (ws + OFF_CSUM);
    float*    Acum   = (float*)   (ws + OFF_ACUM);
    float*    Zbuf   = (float*)   (ws + OFF_Z);
    float*    Gnum   = (float*)   (ws + OFF_GNUM);

    hipMemsetAsync(colsum, 0, 512, stream);

    caps_pack_w<<<256, 256, 0, stream>>>(conv_w, Bpack);
    caps_pack_a<<<MT * 16, 256, 0, stream>>>(inputs, Apack);   // MT*8*32*16/256 = MT*16
    caps_conv_wmma<<<MT, 128, 0, stream>>>(Apack, Bpack, conv_b, caps);
    caps_colsum<<<dim3(266, BB), 256, 0, stream>>>(caps, colsum);

    // iteration 0 (uniform coupling, analytic)
    caps_routing_step<<<1, 128, 0, stream>>>(rw, colsum, Gnum, Zbuf, Acum, out, 0);

    // iteration 1
    hipMemsetAsync(Zbuf, 0, 8704, stream);   // Z (512B) + Gnum (8192B) contiguous
    caps_logits<<<dim3(133, BB), 256, 0, stream>>>(caps, Acum, Zbuf, Gnum);
    caps_routing_step<<<1, 128, 0, stream>>>(rw, colsum, Gnum, Zbuf, Acum, out, 1);

    // iteration 2 (writes output)
    hipMemsetAsync(Zbuf, 0, 8704, stream);
    caps_logits<<<dim3(133, BB), 256, 0, stream>>>(caps, Acum, Zbuf, Gnum);
    caps_routing_step<<<1, 128, 0, stream>>>(rw, colsum, Gnum, Zbuf, Acum, out, 2);
}
